// Mamba3DBlock_35081292874354
// MI455X (gfx1250) — compile-verified, matrix operand order checked by reading, not run on hardware
//
#include <hip/hip_runtime.h>
#include <hip/hip_bf16.h>

// ---------------------------------------------------------------------------
// Mamba-style block for MI455X (gfx1250, wave32, WMMA).
// Pipeline:
//   K0  : f32->f16 weight conversions; x -> xh [B][N][C] f16 (LDS-tiled transpose)
//   K1  : in_proj GEMM (WMMA f16->f32, 4 n-tiles/wave), xp[B][512][N]
//   K2  : depthwise causal conv + SiLU + sigmoid(z); writes xsT/xshT/deltaT
//   K3  : B/C projections (WMMA, shared A+B frags), Bm/Cm [B][N][16]
//   K4a : chunked SSM scan, local pass (64 chunks of 64 steps, LDS-staged Bm/Cm)
//   K4b : carry propagation with A^64
//   K4c : carry correction (LDS-staged Cm), emits yh f16 [B][N][256]
//   K5  : out_proj GEMM (WMMA, 4 n-tiles/wave) + bias + residual -> tmp
//   K6  : LayerNorm over channels -> d_out
// ---------------------------------------------------------------------------

#define BATCH   4
#define DIM     256
#define DI      256
#define NPTS    4096
#define DSTATE  16
#define DCONV   4
#define NCHUNK  64
#define CLEN    64   // NPTS / NCHUNK

typedef _Float16 f16x8  __attribute__((ext_vector_type(8)));
typedef _Float16 f16x16 __attribute__((ext_vector_type(16)));
typedef float    f32x8  __attribute__((ext_vector_type(8)));

// Load one 16x32 f16 WMMA fragment (A or B operand) for k-block `kb`.
// `row` points at the K-contiguous row/column for this lane (lane&15 selects
// the M/N line). Lane-half `hi` = lane>>4 selects K subsets per the ISA
// 16-bit A/B striping.
__device__ inline f16x16 load_frag16(const _Float16* __restrict__ row, int kb, int hi) {
    const f16x8* q0 = (const f16x8*)(row + kb + hi * 8);
    const f16x8* q1 = (const f16x8*)(row + kb + hi * 8 + 16);
    f16x8 lo = *q0;
    f16x8 hi8 = *q1;
    return __builtin_shufflevector(lo, hi8, 0,1,2,3,4,5,6,7,8,9,10,11,12,13,14,15);
}

__device__ inline f32x8 wmma_f16(f16x16 a, f16x16 b, f32x8 c) {
    return __builtin_amdgcn_wmma_f32_16x16x32_f16(
        /*neg_a=*/false, a, /*neg_b=*/false, b,
        /*c_mod=*/(short)0, c, /*reuse_a=*/false, /*reuse_b=*/false);
}

__device__ inline float sigmoidf_dev(float v) { return 1.0f / (1.0f + __expf(-v)); }

// ---------------------------------------------------------------- K0 helpers
__global__ __launch_bounds__(256) void k_conv_f2h(const float* __restrict__ src,
                                                  _Float16* __restrict__ dst, int n) {
    int i = blockIdx.x * 256 + threadIdx.x;
    if (i < n) dst[i] = (_Float16)src[i];
}

// x[B][C][N] f32 -> xh[B][N][C] f16, 32x32 LDS tile transpose (both sides coalesced)
__global__ __launch_bounds__(256) void k_transpose_x(const float* __restrict__ x,
                                                     _Float16* __restrict__ xh) {
    __shared__ float tile[32][33];                // +1 pad: avoid bank conflicts
    int tx = threadIdx.x & 31;
    int ty = threadIdx.x >> 5;                    // 0..7
    int cT = blockIdx.x & 7;                      // DIM/32 = 8
    int nT = (blockIdx.x >> 3) & 127;             // NPTS/32 = 128
    int b  = blockIdx.x >> 10;
    int c0 = cT * 32, n0 = nT * 32;
    #pragma unroll
    for (int j = 0; j < 4; ++j) {
        int c = c0 + ty + j * 8;
        tile[ty + j * 8][tx] = x[((size_t)(b * DIM + c)) * NPTS + n0 + tx];
    }
    __syncthreads();
    #pragma unroll
    for (int j = 0; j < 4; ++j) {
        int n = n0 + ty + j * 8;
        xh[((size_t)b * NPTS + n) * DIM + c0 + tx] = (_Float16)tile[tx][ty + j * 8];
    }
}

// ------------------------------------------------------------ K1: in_proj
// xp[b][o][n] = sum_c w_in[o][c] * x[b][c][n] + b_in[o]
// Each wave: one 16-row A block x 4 consecutive 16-col n-tiles (A-frag reuse x4).
__global__ __launch_bounds__(256) void k_gemm_inproj(const _Float16* __restrict__ whI,
                                                     const _Float16* __restrict__ xh,
                                                     const float* __restrict__ b_in,
                                                     float* __restrict__ xp) {
    int lane = threadIdx.x & 31;
    int wid  = threadIdx.x >> 5;
    int task = blockIdx.x * 8 + wid;              // 4*32*64 = 8192 tasks
    int nG = task & 63;                           // group of 4 n-tiles
    int oT = (task >> 6) & 31;
    int b  = task >> 11;
    int ll = lane & 15, hi = lane >> 4;

    const _Float16* Arow  = whI + (size_t)(oT * 16 + ll) * DIM;
    const _Float16* Bbase = xh + ((size_t)b * NPTS + nG * 64 + ll) * DIM;

    f32x8 acc0 = {0,0,0,0,0,0,0,0}, acc1 = acc0, acc2 = acc0, acc3 = acc0;
    #pragma unroll
    for (int kb = 0; kb < DIM; kb += 32) {
        f16x16 a = load_frag16(Arow, kb, hi);
        acc0 = wmma_f16(a, load_frag16(Bbase,                 kb, hi), acc0);
        acc1 = wmma_f16(a, load_frag16(Bbase + 1 * 16 * DIM,  kb, hi), acc1);
        acc2 = wmma_f16(a, load_frag16(Bbase + 2 * 16 * DIM,  kb, hi), acc2);
        acc3 = wmma_f16(a, load_frag16(Bbase + 3 * 16 * DIM,  kb, hi), acc3);
    }
    int o0 = oT * 16, n0 = nG * 64;
    f32x8 accs[4] = {acc0, acc1, acc2, acc3};
    #pragma unroll
    for (int i = 0; i < 4; ++i) {
        #pragma unroll
        for (int r = 0; r < 8; ++r) {
            int o = o0 + hi * 8 + r;
            xp[((size_t)(b * (2 * DI) + o)) * NPTS + n0 + i * 16 + ll] = accs[i][r] + b_in[o];
        }
    }
}

// ------------------------------------------------- K2: depthwise conv + SiLU
__global__ __launch_bounds__(256) void k_dwconv_silu(const float* __restrict__ xp,
                                                     const float* __restrict__ w_conv,
                                                     const float* __restrict__ b_conv,
                                                     float* __restrict__ xsT,
                                                     _Float16* __restrict__ xshT,
                                                     float* __restrict__ deltaT) {
    int i = blockIdx.x * 256 + threadIdx.x;       // over B*DI*NPTS, n innermost
    int n = i & (NPTS - 1);
    int d = (i >> 12) & (DI - 1);
    int b = i >> 20;
    const float* row = xp + ((size_t)(b * (2 * DI) + d)) * NPTS;
    float acc = b_conv[d];
    #pragma unroll
    for (int j = 0; j < DCONV; ++j) {
        int nn = n - (DCONV - 1) + j;
        if (nn >= 0) acc += w_conv[d * DCONV + j] * row[nn];
    }
    float sil = acc * sigmoidf_dev(acc);
    float z = xp[((size_t)(b * (2 * DI) + DI + d)) * NPTS + n];
    size_t t = ((size_t)(b * NPTS) + n) * DI + d;
    xsT[t]    = sil;
    xshT[t]   = (_Float16)sil;
    deltaT[t] = sigmoidf_dev(z);
}

// -------------------------------------------- K3: B/C projections via WMMA
// Bm[b][n][s] = sum_d xsT[b][n][d] * Bw[s][d]; Cm likewise.
// Each wave: 4 n-tiles sharing the Bw/Cw operand fragments.
__global__ __launch_bounds__(256) void k_gemm_bc(const _Float16* __restrict__ xshT,
                                                 const _Float16* __restrict__ Bwh,
                                                 const _Float16* __restrict__ Cwh,
                                                 float* __restrict__ Bm,
                                                 float* __restrict__ Cm) {
    int lane = threadIdx.x & 31;
    int wid  = threadIdx.x >> 5;
    int task = blockIdx.x * 8 + wid;              // 4*64 = 256 tasks
    int nG = task & 63;
    int b  = task >> 6;
    int ll = lane & 15, hi = lane >> 4;

    const _Float16* Abase = xshT + ((size_t)b * NPTS + nG * 64 + ll) * DI;
    const _Float16* BrowB = Bwh + (size_t)ll * DI;   // column s = ll, K-contig
    const _Float16* BrowC = Cwh + (size_t)ll * DI;

    f32x8 zero = {0,0,0,0,0,0,0,0};
    f32x8 accB[4] = {zero, zero, zero, zero};
    f32x8 accC[4] = {zero, zero, zero, zero};
    #pragma unroll
    for (int kb = 0; kb < DI; kb += 32) {
        f16x16 fb = load_frag16(BrowB, kb, hi);
        f16x16 fc = load_frag16(BrowC, kb, hi);
        #pragma unroll
        for (int i = 0; i < 4; ++i) {
            f16x16 a = load_frag16(Abase + (size_t)i * 16 * DI, kb, hi);
            accB[i] = wmma_f16(a, fb, accB[i]);
            accC[i] = wmma_f16(a, fc, accC[i]);
        }
    }
    #pragma unroll
    for (int i = 0; i < 4; ++i) {
        #pragma unroll
        for (int r = 0; r < 8; ++r) {
            size_t n = (size_t)b * NPTS + nG * 64 + i * 16 + hi * 8 + r;
            Bm[n * DSTATE + ll] = accB[i][r];
            Cm[n * DSTATE + ll] = accC[i][r];
        }
    }
}

// ------------------------------------------- K4a: chunked scan (local pass)
// Bm/Cm for the 64-step chunk are staged once into LDS and read via broadcast.
__global__ __launch_bounds__(256) void k_scan_local(const float* __restrict__ A_log,
                                                    const float* __restrict__ D_skip,
                                                    const float* __restrict__ deltaT,
                                                    const float* __restrict__ xsT,
                                                    const float* __restrict__ Bm,
                                                    const float* __restrict__ Cm,
                                                    float* __restrict__ yBuf,
                                                    float* __restrict__ hfinal) {
    __shared__ float sB[CLEN * DSTATE];           // 4 KB
    __shared__ float sC[CLEN * DSTATE];           // 4 KB
    int b = blockIdx.x >> 6;
    int c = blockIdx.x & (NCHUNK - 1);
    int d = threadIdx.x;
    int n0 = c * CLEN;

    size_t mbase = ((size_t)b * NPTS + n0) * DSTATE;
    #pragma unroll
    for (int i = threadIdx.x; i < CLEN * DSTATE; i += 256) {
        sB[i] = Bm[mbase + i];
        sC[i] = Cm[mbase + i];
    }
    __syncthreads();

    float a[DSTATE], h[DSTATE];
    #pragma unroll
    for (int s = 0; s < DSTATE; ++s) {
        a[s] = -__expf(A_log[d * DSTATE + s]);
        h[s] = 0.0f;
    }
    float Dd = D_skip[d];
    for (int t = 0; t < CLEN; ++t) {
        size_t base = (size_t)b * NPTS + (n0 + t);
        float delta = deltaT[base * DI + d];
        float xv    = xsT[base * DI + d];
        float y = Dd * xv;
        #pragma unroll
        for (int s = 0; s < DSTATE; ++s) {
            h[s] = a[s] * h[s] + delta * sB[t * DSTATE + s];
            y += h[s] * sC[t * DSTATE + s];
        }
        yBuf[base * DI + d] = y;
    }
    size_t hb = (((size_t)blockIdx.x) * DI + d) * DSTATE;   // [b][c][d][s]
    #pragma unroll
    for (int s = 0; s < DSTATE; ++s) hfinal[hb + s] = h[s];
}

// --------------------------------------------- K4b: carry propagation (A^64)
__global__ __launch_bounds__(256) void k_scan_carry(const float* __restrict__ A_log,
                                                    const float* __restrict__ hfinal,
                                                    float* __restrict__ carry) {
    int i = blockIdx.x * 256 + threadIdx.x;       // over B*DI*DSTATE = 16384
    int s = i & (DSTATE - 1);
    int d = (i >> 4) & (DI - 1);
    int b = i >> 12;
    float a = -__expf(A_log[d * DSTATE + s]);
    float ap = a;
    #pragma unroll
    for (int k = 0; k < 6; ++k) ap *= ap;         // a^64
    float cr = 0.0f;
    for (int c = 0; c < NCHUNK; ++c) {
        size_t idx = (((size_t)(b * NCHUNK + c)) * DI + d) * DSTATE + s;
        carry[idx] = cr;                          // state entering chunk c
        cr = hfinal[idx] + ap * cr;
    }
}

// ------------------------------------ K4c: carry correction, emit f16 for K5
__global__ __launch_bounds__(256) void k_scan_fixup(const float* __restrict__ A_log,
                                                    const float* __restrict__ carry,
                                                    const float* __restrict__ Cm,
                                                    const float* __restrict__ yBuf,
                                                    _Float16* __restrict__ yh) {
    __shared__ float sC[CLEN * DSTATE];           // 4 KB
    int b = blockIdx.x >> 6;
    int c = blockIdx.x & (NCHUNK - 1);
    int d = threadIdx.x;
    int n0 = c * CLEN;

    size_t mbase = ((size_t)b * NPTS + n0) * DSTATE;
    #pragma unroll
    for (int i = threadIdx.x; i < CLEN * DSTATE; i += 256) sC[i] = Cm[mbase + i];
    __syncthreads();

    float a[DSTATE], p[DSTATE];
    size_t cb = (((size_t)blockIdx.x) * DI + d) * DSTATE;
    #pragma unroll
    for (int s = 0; s < DSTATE; ++s) {
        a[s] = -__expf(A_log[d * DSTATE + s]);
        p[s] = a[s] * carry[cb + s];              // A^{1} * carry at t=0
    }
    for (int t = 0; t < CLEN; ++t) {
        size_t base = (size_t)b * NPTS + (n0 + t);
        float corr = 0.0f;
        #pragma unroll
        for (int s = 0; s < DSTATE; ++s) corr += p[s] * sC[t * DSTATE + s];
        yh[base * DI + d] = (_Float16)(yBuf[base * DI + d] + corr);
        #pragma unroll
        for (int s = 0; s < DSTATE; ++s) p[s] *= a[s];
    }
}

// ----------------------------------- K5: out_proj GEMM + bias + residual
__global__ __launch_bounds__(256) void k_gemm_outproj(const _Float16* __restrict__ wouth,
                                                      const _Float16* __restrict__ yh,
                                                      const float* __restrict__ b_out,
                                                      const float* __restrict__ x,
                                                      float* __restrict__ tmp) {
    int lane = threadIdx.x & 31;
    int wid  = threadIdx.x >> 5;
    int task = blockIdx.x * 8 + wid;              // 4*16*64 = 4096 tasks
    int nG = task & 63;
    int oT = (task >> 6) & 15;
    int b  = task >> 10;
    int ll = lane & 15, hi = lane >> 4;

    const _Float16* Arow  = wouth + (size_t)(oT * 16 + ll) * DI;
    const _Float16* Bbase = yh + ((size_t)b * NPTS + nG * 64 + ll) * DI;

    f32x8 acc0 = {0,0,0,0,0,0,0,0}, acc1 = acc0, acc2 = acc0, acc3 = acc0;
    #pragma unroll
    for (int kb = 0; kb < DI; kb += 32) {
        f16x16 a = load_frag16(Arow, kb, hi);
        acc0 = wmma_f16(a, load_frag16(Bbase,                kb, hi), acc0);
        acc1 = wmma_f16(a, load_frag16(Bbase + 1 * 16 * DI,  kb, hi), acc1);
        acc2 = wmma_f16(a, load_frag16(Bbase + 2 * 16 * DI,  kb, hi), acc2);
        acc3 = wmma_f16(a, load_frag16(Bbase + 3 * 16 * DI,  kb, hi), acc3);
    }
    int o0 = oT * 16, n0 = nG * 64;
    f32x8 accs[4] = {acc0, acc1, acc2, acc3};
    #pragma unroll
    for (int i = 0; i < 4; ++i) {
        #pragma unroll
        for (int r = 0; r < 8; ++r) {
            int o = o0 + hi * 8 + r;
            size_t idx = ((size_t)(b * DIM + o)) * NPTS + n0 + i * 16 + ll;
            tmp[idx] = accs[i][r] + b_out[o] + x[idx];
        }
    }
}

// --------------------------------------------------- K6: LayerNorm over C
__global__ __launch_bounds__(256) void k_layernorm(const float* __restrict__ tmp,
                                                   const float* __restrict__ gamma,
                                                   const float* __restrict__ beta,
                                                   float* __restrict__ out) {
    int i = blockIdx.x * 256 + threadIdx.x;       // over B*NPTS
    int n = i & (NPTS - 1);
    int b = i >> 12;
    float mean = 0.0f, m2 = 0.0f;
    for (int ch = 0; ch < DIM; ++ch) {
        float v = tmp[((size_t)(b * DIM + ch)) * NPTS + n];
        mean += v; m2 += v * v;
    }
    mean *= (1.0f / DIM);
    float var = m2 * (1.0f / DIM) - mean * mean;
    float rstd = rsqrtf(var + 1e-5f);
    for (int ch = 0; ch < DIM; ++ch) {
        size_t idx = ((size_t)(b * DIM + ch)) * NPTS + n;
        out[idx] = (tmp[idx] - mean) * rstd * gamma[ch] + beta[ch];
    }
}

// ---------------------------------------------------------------------------
extern "C" void kernel_launch(void* const* d_in, const int* in_sizes, int n_in,
                              void* d_out, int out_size, void* d_ws, size_t ws_size,
                              hipStream_t stream) {
    const float* x      = (const float*)d_in[0];
    const float* w_in   = (const float*)d_in[1];
    const float* b_in   = (const float*)d_in[2];
    const float* w_conv = (const float*)d_in[3];
    const float* b_conv = (const float*)d_in[4];
    const float* A_log  = (const float*)d_in[5];
    const float* D_skip = (const float*)d_in[6];
    const float* Bw     = (const float*)d_in[7];
    const float* Cw     = (const float*)d_in[8];
    const float* w_out  = (const float*)d_in[9];
    const float* b_out  = (const float*)d_in[10];
    const float* gamma  = (const float*)d_in[11];
    const float* beta   = (const float*)d_in[12];
    float* out = (float*)d_out;

    // -------- workspace layout (256B-aligned cursor, with buffer reuse) -----
    char* ws = (char*)d_ws;
    size_t off = 0;
    auto alloc = [&](size_t bytes) {
        size_t r = off;
        off = (off + bytes + 255) & ~(size_t)255;
        return r;
    };
    const size_t o_xh     = alloc((size_t)BATCH * NPTS * DIM * 2);       // f16, reused as xshT
    const size_t o_whI    = alloc((size_t)(2 * DI) * DIM * 2);
    const size_t o_Bwh    = alloc((size_t)DSTATE * DIM * 2);
    const size_t o_Cwh    = alloc((size_t)DSTATE * DIM * 2);
    const size_t o_wouth  = alloc((size_t)DIM * DI * 2);
    const size_t o_xp     = alloc((size_t)BATCH * (2 * DI) * NPTS * 4);  // reused: yBuf + tmp
    const size_t o_xsT    = alloc((size_t)BATCH * NPTS * DI * 4);
    const size_t o_deltaT = alloc((size_t)BATCH * NPTS * DI * 4);        // reused as yh (f16)
    const size_t o_Bm     = alloc((size_t)BATCH * NPTS * DSTATE * 4);
    const size_t o_Cm     = alloc((size_t)BATCH * NPTS * DSTATE * 4);
    const size_t o_hfin   = alloc((size_t)BATCH * NCHUNK * DI * DSTATE * 4);
    const size_t o_carry  = alloc((size_t)BATCH * NCHUNK * DI * DSTATE * 4);
    (void)ws_size;

    _Float16* xh     = (_Float16*)(ws + o_xh);
    _Float16* whI    = (_Float16*)(ws + o_whI);
    _Float16* Bwh    = (_Float16*)(ws + o_Bwh);
    _Float16* Cwh    = (_Float16*)(ws + o_Cwh);
    _Float16* wouth  = (_Float16*)(ws + o_wouth);
    float*    xp     = (float*)(ws + o_xp);
    float*    yBuf   = xp;                                               // xp dead after K2
    float*    tmp    = xp + (size_t)BATCH * NPTS * DI;                   // second half of xp
    float*    xsT    = (float*)(ws + o_xsT);
    _Float16* xshT   = xh;                                               // xh dead after K1
    float*    deltaT = (float*)(ws + o_deltaT);
    _Float16* yh     = (_Float16*)(ws + o_deltaT);                       // deltaT dead after K4a
    float*    Bm     = (float*)(ws + o_Bm);
    float*    Cm     = (float*)(ws + o_Cm);
    float*    hfinal = (float*)(ws + o_hfin);
    float*    carry  = (float*)(ws + o_carry);

    // K0: conversions
    k_conv_f2h<<<(2 * DI * DIM + 255) / 256, 256, 0, stream>>>(w_in, whI, 2 * DI * DIM);
    k_conv_f2h<<<(DSTATE * DIM + 255) / 256, 256, 0, stream>>>(Bw, Bwh, DSTATE * DIM);
    k_conv_f2h<<<(DSTATE * DIM + 255) / 256, 256, 0, stream>>>(Cw, Cwh, DSTATE * DIM);
    k_conv_f2h<<<(DIM * DI + 255) / 256, 256, 0, stream>>>(w_out, wouth, DIM * DI);
    k_transpose_x<<<BATCH * (NPTS / 32) * (DIM / 32), 256, 0, stream>>>(x, xh);

    // K1: in_proj (WMMA, 4 n-tiles per wave)
    k_gemm_inproj<<<(BATCH * 32 * 64) / 8, 256, 0, stream>>>(whI, xh, b_in, xp);

    // K2: depthwise conv + SiLU + delta
    k_dwconv_silu<<<(BATCH * DI * NPTS) / 256, 256, 0, stream>>>(
        xp, w_conv, b_conv, xsT, xshT, deltaT);

    // K3: B/C projections (WMMA)
    k_gemm_bc<<<(BATCH * 64) / 8, 256, 0, stream>>>(xshT, Bwh, Cwh, Bm, Cm);

    // K4: chunked SSM scan
    k_scan_local<<<BATCH * NCHUNK, 256, 0, stream>>>(
        A_log, D_skip, deltaT, xsT, Bm, Cm, yBuf, hfinal);
    k_scan_carry<<<(BATCH * DI * DSTATE) / 256, 256, 0, stream>>>(A_log, hfinal, carry);
    k_scan_fixup<<<BATCH * NCHUNK, 256, 0, stream>>>(A_log, carry, Cm, yBuf, yh);

    // K5: out_proj (WMMA, 4 n-tiles per wave) + residual
    k_gemm_outproj<<<(BATCH * 16 * 64) / 8, 256, 0, stream>>>(wouth, yh, b_out, x, tmp);

    // K6: LayerNorm over channels
    k_layernorm<<<(BATCH * NPTS) / 256, 256, 0, stream>>>(tmp, gamma, beta, out);
}
